// WindowAttention_37349035606825
// MI455X (gfx1250) — matrix operand order repaired and checked by reading, not compiled, not run on hardware
//
#include <hip/hip_runtime.h>
#include <hip/hip_bf16.h>
#include <math.h>

typedef _Float16 f16;
typedef __attribute__((ext_vector_type(16))) _Float16 v16h;
typedef __attribute__((ext_vector_type(8)))  float    v8f;
typedef __attribute__((ext_vector_type(4)))  unsigned int u32x4;
typedef __attribute__((ext_vector_type(4)))  int i32x4;
typedef __attribute__((ext_vector_type(8)))  int i32x8;

#define NTOK  144
#define DIMC  512
#define HEADS 16
#define HD    32
#define BWIN  256
#define NWIN  64

union HFrag { v16h v; unsigned u[8]; };

__device__ __forceinline__ unsigned pack_f2(float a, float b) {
  union { _Float16 h[2]; unsigned u; } t;
  t.h[0] = (_Float16)a; t.h[1] = (_Float16)b; return t.u;
}
__device__ __forceinline__ unsigned pack_h2(f16 a, f16 b) {
  union { _Float16 h[2]; unsigned u; } t;
  t.h[0] = a; t.h[1] = b; return t.u;
}
// A-fragment K offset for 16x32 f16 A matrix (contiguous: k = 16*hi + 2*j)
__device__ __forceinline__ int koffA(int j, int hi) {
  return 2 * j + hi * 16;
}

// raw 32-bit LDS byte offset of a generic pointer into shared memory
typedef __attribute__((address_space(3))) char lds_char_t;
__device__ __forceinline__ unsigned lds_addr_of(void* p) {
  return (unsigned)(unsigned long long)(lds_char_t*)p;
}

// issue one TDM 2-D tile load: rows x (rowbytes/8) units of 8B, optional LDS row pad
__device__ __forceinline__ void tdm_load_2d(unsigned lds_off, const void* gptr,
                                            unsigned units_per_row, unsigned rows,
                                            unsigned pad_ctl /* bits20..31 of g1.w0 */) {
  unsigned long long ga = (unsigned long long)gptr;
  u32x4 g0 = { 1u /*count=1*/, lds_off, (unsigned)ga,
               ((unsigned)(ga >> 32) & 0x1FFFFFFu) | (2u << 30) /*type=2*/ };
  i32x8 g1 = { (int)((3u << 16) | pad_ctl),        // data_size=8B, pad controls
               (int)(units_per_row << 16),         // tensor_dim0[15:0]
               (int)(rows << 16),                  // tensor_dim1[15:0] (dim0 hi = 0)
               (int)(units_per_row << 16),         // tile_dim0  (dim1 hi = 0)
               (int)rows,                          // tile_dim1, tile_dim2=0
               (int)units_per_row,                 // tensor_dim0_stride[31:0]
               0, 0 };
  i32x4 z4 = { 0, 0, 0, 0 };
#if defined(__clang_major__) && (__clang_major__ >= 23)
  i32x8 z8 = { 0, 0, 0, 0, 0, 0, 0, 0 };
  __builtin_amdgcn_tensor_load_to_lds(g0, g1, z4, z4, z8, 0);
#else
  __builtin_amdgcn_tensor_load_to_lds(g0, g1, z4, z4, 0);
#endif
}

// ---------------- f32 -> f16 conversion (weights) ----------------
__global__ void cvt_f32_f16(const float* __restrict__ in, f16* __restrict__ out, int n) {
  int i = blockIdx.x * blockDim.x + threadIdx.x;
  if (i < n) out[i] = (f16)in[i];
}

// ---------------- CPB MLP: tbl[529][16] ----------------
__global__ void cpb_kernel(const float* __restrict__ rel_table,
                           const float* __restrict__ w1, const float* __restrict__ b1,
                           const float* __restrict__ w2, float* __restrict__ tbl) {
  __shared__ float acc[HEADS];
  const int r = blockIdx.x, tid = threadIdx.x;
  if (tid < HEADS) acc[tid] = 0.f;
  __syncthreads();
  const float t0 = rel_table[2 * r], t1 = rel_table[2 * r + 1];
  float part[HEADS];
#pragma unroll
  for (int h = 0; h < HEADS; ++h) part[h] = 0.f;
  for (int c = tid; c < 512; c += blockDim.x) {
    float hv = fmaxf(t0 * w1[2 * c] + t1 * w1[2 * c + 1] + b1[c], 0.f);
#pragma unroll
    for (int h = 0; h < HEADS; ++h) part[h] += hv * w2[h * 512 + c];
  }
#pragma unroll
  for (int h = 0; h < HEADS; ++h) atomicAdd(&acc[h], part[h]);
  __syncthreads();
  if (tid < HEADS) tbl[r * HEADS + tid] = acc[tid];
}

// ---------------- QKV GEMM: (36864x512) x (1536x512)^T ----------------
__global__ __launch_bounds__(256) void qkv_gemm(
    const float* __restrict__ x, const f16* __restrict__ wq,
    const float* __restrict__ bias,
    f16* __restrict__ qb, f16* __restrict__ kb, f16* __restrict__ vb) {
  const int lane = threadIdx.x, wave = threadIdx.y;
  const int hi = lane >> 4, lr = lane & 15;
  const int tileM = blockIdx.y * 8 + wave;
  const int col0 = blockIdx.x * 64;
  const float* xrow = x + (size_t)(tileM * 16 + lr) * DIMC;

  v8f acc[4];
#pragma unroll
  for (int g = 0; g < 4; ++g) acc[g] = (v8f){};

  for (int kk = 0; kk < DIMC; kk += 32) {
    __builtin_prefetch(xrow + kk + 64, 0, 1);  // global_prefetch_b8 next K slab
    HFrag a;
#pragma unroll
    for (int j = 0; j < 8; ++j) {
      float2 f = *(const float2*)(xrow + kk + koffA(j, hi));
      a.u[j] = pack_f2(f.x, f.y);
    }
#pragma unroll
    for (int g = 0; g < 4; ++g) {
      const f16* wrow = wq + (size_t)(col0 + g * 16 + lr) * DIMC + kk + hi * 16;
      HFrag b;
#pragma unroll
      for (int j = 0; j < 8; ++j) b.u[j] = *(const unsigned*)(wrow + 2 * j);
      acc[g] = __builtin_amdgcn_wmma_f32_16x16x32_f16(false, a.v, false, b.v,
                                                      (short)0, acc[g], false, false);
    }
  }
  const float qscale = 0.17677669529663687f;  // 32^-0.5, folded into q
#pragma unroll
  for (int r = 0; r < 8; ++r) {
    int m = tileM * 16 + r + hi * 8;
    int bw = m / NTOK, n = m % NTOK;
#pragma unroll
    for (int g = 0; g < 4; ++g) {
      int c = col0 + g * 16 + lr;
      int s = c >> 9, ch = c & 511;
      int h = ch >> 5, d = ch & 31;
      float v = acc[g][r] + bias[c];
      if (s == 0) v *= qscale;
      f16* dst = (s == 0) ? qb : (s == 1) ? kb : vb;
      dst[(((size_t)bw * HEADS + h) * NTOK + n) * HD + d] = (f16)v;
    }
  }
}

// ---------------- fused attention: S=QK^T, +bias+mask, softmax, O=PV ----------------
// block = (32,9): one wave per 16-row strip of S; one block per (b,h)
// K/V staged into LDS by the Tensor Data Mover (TDM); V row-padded to 40 halfwords
// in hardware via D# pad_interval/pad_amount (pad 4 dwords every 16 dwords).
__global__ __launch_bounds__(288) void attn_kernel(
    const f16* __restrict__ qb, const f16* __restrict__ kb, const f16* __restrict__ vb,
    const float* __restrict__ tbl, const int* __restrict__ ridx,
    const float* __restrict__ mask, f16* __restrict__ ob) {
  extern __shared__ char smem[];
  f16* P  = (f16*)smem;                               // 9 waves * 16 x 160 (K padded)
  f16* kS = (f16*)(smem + 9 * 16 * 160 * 2);          // 144 x 32
  f16* vS = (f16*)(smem + 9 * 16 * 160 * 2 + NTOK * HD * 2); // 160 x 40 (row pad)

  const int lane = threadIdx.x, wave = threadIdx.y;
  const int tid = wave * 32 + lane;
  const int bh = blockIdx.x, b = bh >> 4, h = bh & 15;
  const size_t base = ((size_t)b * HEADS + h) * (NTOK * HD);

  if (wave == 0) {
    // K: 144 rows x 8 x 8B, contiguous in LDS
    tdm_load_2d(lds_addr_of(kS), kb + base, 8u, (unsigned)NTOK, 0u);
    // V: 144 rows x 8 x 8B, +16B LDS pad per row (pad_en, interval=16dw, amount=4dw)
    tdm_load_2d(lds_addr_of(vS), vb + base, 8u, (unsigned)NTOK,
                (1u << 4) | (3u << 6) | (3u << 9));  // bits20,24:22,31:25 of g1.w0 >> 16
    __builtin_amdgcn_s_wait_tensorcnt(0);
  }
  // zero-fill V pad rows 144..159 and P K-pad cols 144..159
  for (int e = tid; e < 16 * HD; e += 288)  { int r = 144 + (e >> 5), c = e & 31; vS[r * 40 + c] = (f16)0.f; }
  for (int e = lane; e < 16 * 16; e += 32)  { int r = e >> 4, c = 144 + (e & 15); P[wave * 2560 + r * 160 + c] = (f16)0.f; }
  __syncthreads();

  const int hi = lane >> 4, lr = lane & 15;
  HFrag a;
  {
    const f16* qrow = qb + base + (size_t)(wave * 16 + lr) * HD;
#pragma unroll
    for (int j = 0; j < 8; ++j) a.u[j] = *(const unsigned*)(qrow + koffA(j, hi));
  }
  const float* maskp = mask + (size_t)(b & (NWIN - 1)) * NTOK * NTOK;

  v8f S[9];
#pragma unroll
  for (int jt = 0; jt < 9; ++jt) {
    HFrag bf;
    const f16* krow = kS + (jt * 16 + lr) * HD + hi * 16;
#pragma unroll
    for (int j = 0; j < 8; ++j) bf.u[j] = *(const unsigned*)(krow + 2 * j);
    v8f c = {};
    c = __builtin_amdgcn_wmma_f32_16x16x32_f16(false, a.v, false, bf.v, (short)0, c, false, false);
#pragma unroll
    for (int r = 0; r < 8; ++r) {
      int mg = wave * 16 + r + hi * 8;
      int ng = jt * 16 + lr;
      c[r] += tbl[ridx[mg * NTOK + ng] * HEADS + h] + maskp[mg * NTOK + ng];
    }
    S[jt] = c;
  }

  // softmax over 144 cols: rows live in 16-lane halves -> xor-shuffle reduce
#pragma unroll
  for (int r = 0; r < 8; ++r) {
    float mx = -1e30f;
#pragma unroll
    for (int jt = 0; jt < 9; ++jt) mx = fmaxf(mx, S[jt][r]);
#pragma unroll
    for (int o = 8; o >= 1; o >>= 1) mx = fmaxf(mx, __shfl_xor(mx, o, 32));
    float sum = 0.f;
#pragma unroll
    for (int jt = 0; jt < 9; ++jt) { float e = __expf(S[jt][r] - mx); S[jt][r] = e; sum += e; }
#pragma unroll
    for (int o = 8; o >= 1; o >>= 1) sum += __shfl_xor(sum, o, 32);
    float inv = 1.f / sum;
    f16* prow = P + wave * 2560 + (r + hi * 8) * 160;
#pragma unroll
    for (int jt = 0; jt < 9; ++jt) prow[jt * 16 + lr] = (f16)(S[jt][r] * inv);
  }
  __syncthreads();

  // O = P(16x160) @ V(160x32), two 16x16 N-tiles, K in 5 steps of 32
  v8f o0 = {}, o1 = {};
#pragma unroll
  for (int kc = 0; kc < 5; ++kc) {
    HFrag pa;
    const f16* prow = P + wave * 2560 + lr * 160 + kc * 32;
#pragma unroll
    for (int j = 0; j < 8; ++j) pa.u[j] = *(const unsigned*)(prow + koffA(j, hi));
    HFrag b0, b1;
#pragma unroll
    for (int j = 0; j < 8; ++j) {
      int kr = kc * 32 + hi * 16 + 2 * j;
      b0.u[j] = pack_h2(vS[kr * 40 + lr],      vS[(kr + 1) * 40 + lr]);
      b1.u[j] = pack_h2(vS[kr * 40 + 16 + lr], vS[(kr + 1) * 40 + 16 + lr]);
    }
    o0 = __builtin_amdgcn_wmma_f32_16x16x32_f16(false, pa.v, false, b0.v, (short)0, o0, false, false);
    o1 = __builtin_amdgcn_wmma_f32_16x16x32_f16(false, pa.v, false, b1.v, (short)0, o1, false, false);
  }
#pragma unroll
  for (int r = 0; r < 8; ++r) {
    int mg = wave * 16 + r + hi * 8;
    size_t orow = ((size_t)b * NTOK + mg) * DIMC + h * HD;
    ob[orow + lr]      = (f16)o0[r];
    ob[orow + 16 + lr] = (f16)o1[r];
  }
}

// ---------------- proj GEMM: (36864x512) x (512x512)^T -> f32 out ----------------
__global__ __launch_bounds__(256) void proj_gemm(
    const f16* __restrict__ A, const f16* __restrict__ W,
    const float* __restrict__ bias, float* __restrict__ out) {
  const int lane = threadIdx.x, wave = threadIdx.y;
  const int hi = lane >> 4, lr = lane & 15;
  const int tileM = blockIdx.y * 8 + wave;
  const int col0 = blockIdx.x * 64;
  const f16* arow = A + (size_t)(tileM * 16 + lr) * DIMC;

  v8f acc[4];
#pragma unroll
  for (int g = 0; g < 4; ++g) acc[g] = (v8f){};

  for (int kk = 0; kk < DIMC; kk += 32) {
    __builtin_prefetch(arow + kk + 64, 0, 1);
    HFrag a;
#pragma unroll
    for (int j = 0; j < 8; ++j) a.u[j] = *(const unsigned*)(arow + kk + koffA(j, hi));
#pragma unroll
    for (int g = 0; g < 4; ++g) {
      const f16* wrow = W + (size_t)(col0 + g * 16 + lr) * DIMC + kk + hi * 16;
      HFrag b;
#pragma unroll
      for (int j = 0; j < 8; ++j) b.u[j] = *(const unsigned*)(wrow + 2 * j);
      acc[g] = __builtin_amdgcn_wmma_f32_16x16x32_f16(false, a.v, false, b.v,
                                                      (short)0, acc[g], false, false);
    }
  }
#pragma unroll
  for (int r = 0; r < 8; ++r) {
    size_t m = (size_t)tileM * 16 + r + hi * 8;
#pragma unroll
    for (int g = 0; g < 4; ++g) {
      int c = col0 + g * 16 + lr;
      out[m * DIMC + c] = acc[g][r] + bias[c];
    }
  }
}

extern "C" void kernel_launch(void* const* d_in, const int* in_sizes, int n_in,
                              void* d_out, int out_size, void* d_ws, size_t ws_size,
                              hipStream_t stream) {
  (void)in_sizes; (void)n_in; (void)out_size; (void)ws_size;
  const float* x      = (const float*)d_in[0];
  const float* mask   = (const float*)d_in[1];
  const float* qkv_w  = (const float*)d_in[2];
  const float* qkv_b  = (const float*)d_in[3];
  const float* proj_w = (const float*)d_in[4];
  const float* proj_b = (const float*)d_in[5];
  const float* cpb_w1 = (const float*)d_in[6];
  const float* cpb_b1 = (const float*)d_in[7];
  const float* cpb_w2 = (const float*)d_in[8];
  const float* rel_t  = (const float*)d_in[9];
  const int*   rel_i  = (const int*)d_in[10];
  float* out = (float*)d_out;

  char* ws = (char*)d_ws;
  size_t off = 0;
  f16* qw_h = (f16*)(ws + off); off += (size_t)1536 * 512 * 2;
  f16* pw_h = (f16*)(ws + off); off += (size_t)512 * 512 * 2;
  const size_t qkvN = (size_t)BWIN * HEADS * NTOK * HD;  // 18,874,368
  f16* qb = (f16*)(ws + off); off += qkvN * 2;
  f16* kb = (f16*)(ws + off); off += qkvN * 2;
  f16* vb = (f16*)(ws + off); off += qkvN * 2;
  f16* ob = (f16*)(ws + off); off += qkvN * 2;
  float* tbl = (float*)(ws + off); off += (size_t)529 * HEADS * 4;

  cvt_f32_f16<<<(1536 * 512 + 255) / 256, 256, 0, stream>>>(qkv_w, qw_h, 1536 * 512);
  cvt_f32_f16<<<(512 * 512 + 255) / 256, 256, 0, stream>>>(proj_w, pw_h, 512 * 512);
  cpb_kernel<<<529, 256, 0, stream>>>(rel_t, cpb_w1, cpb_b1, cpb_w2, tbl);
  qkv_gemm<<<dim3(24, 288), dim3(32, 8), 0, stream>>>(x, qw_h, qkv_b, qb, kb, vb);
  // dynamic LDS: P 46080 + K 9216 + V 12800 = 68096 bytes (within 320KB WGP LDS)
  attn_kernel<<<BWIN * HEADS, dim3(32, 9), 68096, stream>>>(qb, kb, vb, tbl, rel_i, mask, ob);
  proj_gemm<<<dim3(8, 288), dim3(32, 8), 0, stream>>>(ob, pw_h, proj_b, out);
}
